// SogCLR_DRO_Loss_36378372997155
// MI455X (gfx1250) — compile-verified, matrix operand order checked by reading, not run on hardware
//
#include <hip/hip_runtime.h>
#include <hip/hip_bf16.h>
#include <stddef.h>

// ---------------------------------------------------------------------------
// Problem constants (match reference)
// ---------------------------------------------------------------------------
#define BSZ      4096
#define TWO_B    8192
#define D_DIM    2048
#define NUM_NEG  8190.0f
#define GAMMA_C  0.8f
#define RHO_C    0.8f
#define ETA_C    0.001f
#define BETAU_C  0.9f
#define CLIP_C   3.0f
#define TAU_MIN_C 0.05f
#define TAU_MAX_C 1.0f

typedef __attribute__((ext_vector_type(16))) __bf16 v16bf;
typedef __attribute__((ext_vector_type(8)))  float  v8f;

// Tile geometry: workgroup = 8 waves, block = 128 rows x 128 cols, K chunk 32.
#define KC       32
#define MB       128
#define LSTRIDE  40                 // bf16 elems per LDS row (80B, conflict-free)
#define ARR      (MB * LSTRIDE)     // elems per LDS sub-array
#define NBUF     3

// gfx1250 async DMA: global -> LDS, 16B per lane, tracked by ASYNCcnt.
#define ASYNC_B128(ldsoff, gptr)                                          \
    asm volatile("global_load_async_to_lds_b128 %0, %1, off"              \
                 :: "v"((unsigned)(ldsoff)), "v"((const void*)(gptr))     \
                 : "memory")

#define WAIT_ASYNC_8()  asm volatile("s_wait_asynccnt 0x8" ::: "memory")
#define WAIT_ASYNC_0()  asm volatile("s_wait_asynccnt 0x0" ::: "memory")

// ---------------------------------------------------------------------------
// 1) Row-normalize features, split into bf16 hi/lo for 3xBF16 fp32 emulation
// ---------------------------------------------------------------------------
__global__ __launch_bounds__(256)
void norm_split_kernel(const float* __restrict__ feat,
                       __bf16* __restrict__ fh, __bf16* __restrict__ fl) {
    const int row = blockIdx.x;
    const int tid = threadIdx.x;
    __shared__ float red[256];
    const float* fr = feat + (size_t)row * D_DIM;
    float s = 0.0f;
    for (int i = tid; i < D_DIM; i += 256) { float v = fr[i]; s += v * v; }
    red[tid] = s;
    __syncthreads();
    for (int off = 128; off > 0; off >>= 1) {
        if (tid < off) red[tid] += red[tid + off];
        __syncthreads();
    }
    const float inv = 1.0f / fmaxf(sqrtf(red[0]), 1e-12f);
    for (int i = tid; i < D_DIM; i += 256) {
        float v = fr[i] * inv;
        __bf16 h = (__bf16)v;
        fh[(size_t)row * D_DIM + i] = h;
        fl[(size_t)row * D_DIM + i] = (__bf16)(v - (float)h);
    }
}

// ---------------------------------------------------------------------------
// 2) Zero row-accumulator arrays
// ---------------------------------------------------------------------------
__global__ void zero_kernel(float* __restrict__ p, int n) {
    int i = blockIdx.x * blockDim.x + threadIdx.x;
    if (i < n) p[i] = 0.0f;
}

// ---------------------------------------------------------------------------
// Fragment load from LDS (16-bit A/B layout: lanes<16 hold K {0..7,16..23},
// lanes>=16 hold K {8..15,24..31} of row/col (lane&15))
// ---------------------------------------------------------------------------
__device__ __forceinline__ v16bf load_frag(const __bf16* base, int half) {
    v16bf f;
    const __bf16* p0 = base + half * 8;
    const __bf16* p1 = base + 16 + half * 8;
#pragma unroll
    for (int i = 0; i < 8; ++i) f[i] = p0[i];
#pragma unroll
    for (int i = 0; i < 8; ++i) f[8 + i] = p1[i];
    return f;
}

// ---------------------------------------------------------------------------
// 3) Fused sim-GEMM (3xBF16 WMMA) + masked exp row-reductions.
//    grid.x = 64 row blocks (128 rows), grid.y = 8 column splits (1024 cols).
//    8 waves / workgroup; wave owns 16 rows x 128 cols (8 C-tiles).
//    3-deep async global->LDS pipeline, one barrier per K-chunk.
// ---------------------------------------------------------------------------
__global__ __launch_bounds__(256)
void fused_sim_kernel(const __bf16* __restrict__ fh, const __bf16* __restrict__ fl,
                      const float* __restrict__ tau_buf, const int* __restrict__ index,
                      float* __restrict__ sum_e, float* __restrict__ sum_es,
                      float* __restrict__ pos_out) {
    const int rowbase  = blockIdx.x * MB;
    const int colsplit = blockIdx.y;             // 0..7, each covers 1024 cols
    const int tid  = threadIdx.x;
    const int wave = tid >> 5;
    const int lane = tid & 31;
    const int half = lane >> 4;
    const int ln   = lane & 15;
    const int wrow = rowbase + wave * 16;        // this wave's 16 rows

    // 3 pipeline buffers x {Ah, Al, Bh, Bl} x (128 x 40) bf16  = 120 KB LDS
    __shared__ __bf16 lds[NBUF][4][ARR];

    // staging role of this thread: 2 threads per staged row, 32B each
    const int lrow  = tid >> 1;                  // 0..127
    const int lpart = tid & 1;

    // per-row 1/tau (row i uses tau_buf[index[i mod B]])
    float itau[8];
#pragma unroll
    for (int r = 0; r < 8; ++r) {
        int grow = wrow + r + half * 8;
        itau[r] = 1.0f / tau_buf[index[grow & (BSZ - 1)]];
    }

    float acc_e[8], acc_es[8], acc_p[8];
#pragma unroll
    for (int r = 0; r < 8; ++r) { acc_e[r] = 0.f; acc_es[r] = 0.f; acc_p[r] = 0.f; }

    const int NKC    = D_DIM / KC;               // 64 K-chunks per col block
    const int NSTAGE = 8 * NKC;                  // 8 col blocks per split

    // issue one stage (8 async b128 ops per wave) into buffer `buf`
    auto stage = [&](int s, int buf) {
        const int cb = s >> 6;
        const int kc = (s & (NKC - 1)) * KC;
        const int colbase = colsplit * 1024 + cb * MB;
        const size_t ga = (size_t)(rowbase + lrow) * D_DIM + kc + lpart * 16;
        const size_t gb = (size_t)(colbase + lrow) * D_DIM + kc + lpart * 16;
        const unsigned lo = (unsigned)((lrow * LSTRIDE + lpart * 16) * 2);
        const unsigned bAh = (unsigned)(uintptr_t)&lds[buf][0][0];
        const unsigned bAl = (unsigned)(uintptr_t)&lds[buf][1][0];
        const unsigned bBh = (unsigned)(uintptr_t)&lds[buf][2][0];
        const unsigned bBl = (unsigned)(uintptr_t)&lds[buf][3][0];
        ASYNC_B128(bAh + lo,      fh + ga);
        ASYNC_B128(bAh + lo + 16, fh + ga + 8);
        ASYNC_B128(bAl + lo,      fl + ga);
        ASYNC_B128(bAl + lo + 16, fl + ga + 8);
        ASYNC_B128(bBh + lo,      fh + gb);
        ASYNC_B128(bBh + lo + 16, fh + gb + 8);
        ASYNC_B128(bBl + lo,      fl + gb);
        ASYNC_B128(bBl + lo + 16, fl + gb + 8);
    };

    v8f c[8];

    stage(0, 0);
    stage(1, 1);

    int buf = 0;      // buffer for compute(s)
    int sbuf = 2;     // buffer for stage(s+2)
    for (int s = 0; s < NSTAGE; ++s) {
        if ((s & (NKC - 1)) == 0) {
#pragma unroll
            for (int n = 0; n < 8; ++n) c[n] = (v8f){0,0,0,0,0,0,0,0};
        }

        // stage(s) complete (only stage(s+1)'s 8 ops may remain in flight)
        if (s < NSTAGE - 1) { WAIT_ASYNC_8(); } else { WAIT_ASYNC_0(); }
        __syncthreads();

        if (s + 2 < NSTAGE) stage(s + 2, sbuf);

        const __bf16* bp = &lds[buf][0][0];
        const v16bf ah = load_frag(bp + 0 * ARR + (wave * 16 + ln) * LSTRIDE, half);
        const v16bf al = load_frag(bp + 1 * ARR + (wave * 16 + ln) * LSTRIDE, half);
#pragma unroll
        for (int n = 0; n < 8; ++n) {
            const v16bf bh = load_frag(bp + 2 * ARR + (n * 16 + ln) * LSTRIDE, half);
            const v16bf bl = load_frag(bp + 3 * ARR + (n * 16 + ln) * LSTRIDE, half);
            // sim ~= hi*hi + hi*lo + lo*hi (fp32-accuracy bf16 emulation)
            c[n] = __builtin_amdgcn_wmma_f32_16x16x32_bf16(
                       false, ah, false, bh, (short)0, c[n], false, false);
            c[n] = __builtin_amdgcn_wmma_f32_16x16x32_bf16(
                       false, ah, false, bl, (short)0, c[n], false, false);
            c[n] = __builtin_amdgcn_wmma_f32_16x16x32_bf16(
                       false, al, false, bh, (short)0, c[n], false, false);
        }

        // end of a col block: masked exp + row-partial accumulation
        if ((s & (NKC - 1)) == (NKC - 1)) {
            const int colbase = colsplit * 1024 + (s >> 6) * MB;
#pragma unroll
            for (int n = 0; n < 8; ++n) {
                const int gcol = colbase + n * 16 + ln;
#pragma unroll
                for (int r = 0; r < 8; ++r) {
                    const int grow = wrow + r + half * 8;
                    const float sim = c[n][r];
                    const bool masked = (((gcol ^ grow) & (BSZ - 1)) == 0);
                    const float e = masked ? 0.0f : __expf(sim * itau[r]);
                    acc_e[r]  += e;
                    acc_es[r] += e * sim;
                    if (gcol == (grow ^ BSZ)) acc_p[r] += sim;  // positive pair
                }
            }
        }

        buf  = (buf  == NBUF - 1) ? 0 : buf + 1;
        sbuf = (sbuf == NBUF - 1) ? 0 : sbuf + 1;
    }

    // 16-lane butterfly reduction (halves reduce independently, masks < 16)
#pragma unroll
    for (int r = 0; r < 8; ++r) {
#pragma unroll
        for (int m = 1; m < 16; m <<= 1) {
            acc_e[r]  += __shfl_xor(acc_e[r],  m, 32);
            acc_es[r] += __shfl_xor(acc_es[r], m, 32);
            acc_p[r]  += __shfl_xor(acc_p[r],  m, 32);
        }
    }
    if (ln == 0) {
#pragma unroll
        for (int r = 0; r < 8; ++r) {
            const int grow = wrow + r + half * 8;
            atomicAdd(&sum_e[grow],   acc_e[r]);
            atomicAdd(&sum_es[grow],  acc_es[r]);
            atomicAdd(&pos_out[grow], acc_p[r]);
        }
    }
}

// ---------------------------------------------------------------------------
// 4) Copy s / u / tau buffers into the output layout
//    d_out = [loss(1) | s_new(N) | u_new(N) | tau_new(N)]
// ---------------------------------------------------------------------------
__global__ void copy_out_kernel(const float* __restrict__ s,
                                const float* __restrict__ u,
                                const float* __restrict__ t,
                                float* __restrict__ out, int N) {
    float* os = out + 1;
    float* ou = os + N;
    float* ot = ou + N;
    const int stride = gridDim.x * blockDim.x;
    for (int i = blockIdx.x * blockDim.x + threadIdx.x; i < N; i += stride) {
        os[i] = s[i]; ou[i] = u[i]; ot[i] = t[i];
    }
}

// ---------------------------------------------------------------------------
// 5) Finalize: per-sample EMA, loss mean, grad_tau, scatter updates
// ---------------------------------------------------------------------------
__global__ __launch_bounds__(1024)
void finalize_kernel(const float* __restrict__ sum_e, const float* __restrict__ sum_es,
                     const float* __restrict__ pos,
                     const float* __restrict__ s_in, const float* __restrict__ tau_in,
                     const float* __restrict__ u_in, const int* __restrict__ index,
                     const int* __restrict__ epoch, float* __restrict__ out, int N) {
    __shared__ float red[1024];
    const int tid = threadIdx.x;
    const int ep = epoch[0];
    float* out_s = out + 1;
    float* out_u = out_s + N;
    float* out_t = out_u + N;
    float lsum = 0.0f;
    for (int i = tid; i < BSZ; i += 1024) {
        const int idx = index[i];
        const float tau  = tau_in[idx];
        const float itau = 1.0f / tau;
        const float g1 = sum_e[i]       * (1.0f / NUM_NEG);
        const float g2 = sum_e[i + BSZ] * (1.0f / NUM_NEG);
        float s1, s2;
        if (ep == 0) { s1 = g1; s2 = g2; }
        else {
            const float so = s_in[idx];
            s1 = (1.0f - GAMMA_C) * so + GAMMA_C * g1;
            s2 = (1.0f - GAMMA_C) * so + GAMMA_C * g2;
        }
        const float inv1 = 1.0f / (s1 * NUM_NEG);
        const float inv2 = 1.0f / (s2 * NUM_NEG);
        lsum += (sum_es[i] * inv1 - pos[i]) + (sum_es[i + BSZ] * inv2 - pos[i + BSZ]);
        const float gt1 = __logf(s1) + RHO_C - sum_es[i]       * itau * inv1;
        const float gt2 = __logf(s2) + RHO_C - sum_es[i + BSZ] * itau * inv2;
        float gt = 0.5f * (gt1 + gt2);
        gt = fminf(fmaxf(gt, -CLIP_C), CLIP_C);
        const float un = (1.0f - BETAU_C) * u_in[idx] + BETAU_C * gt;
        const float tn = fminf(fmaxf(tau - ETA_C * un, TAU_MIN_C), TAU_MAX_C);
        out_s[idx] = 0.5f * (s1 + s2);
        out_u[idx] = un;
        out_t[idx] = tn;
    }
    red[tid] = lsum;
    __syncthreads();
    for (int off = 512; off > 0; off >>= 1) {
        if (tid < off) red[tid] += red[tid + off];
        __syncthreads();
    }
    if (tid == 0) out[0] = red[0] / (float)BSZ;
}

// ---------------------------------------------------------------------------
// Launcher
// ---------------------------------------------------------------------------
extern "C" void kernel_launch(void* const* d_in, const int* in_sizes, int n_in,
                              void* d_out, int out_size, void* d_ws, size_t ws_size,
                              hipStream_t stream) {
    const float* features = (const float*)d_in[0];
    const float* s_buf    = (const float*)d_in[1];
    const float* tau_buf  = (const float*)d_in[2];
    const float* u_buf    = (const float*)d_in[3];
    const int*   index    = (const int*)d_in[4];
    const int*   epoch    = (const int*)d_in[5];
    float*       out      = (float*)d_out;
    const int N = in_sizes[1];

    // workspace layout
    char* ws = (char*)d_ws;
    __bf16* fh = (__bf16*)ws;                                       // 32 MB
    __bf16* fl = (__bf16*)(ws + (size_t)TWO_B * D_DIM * 2);         // 32 MB
    float* sum_e  = (float*)(ws + (size_t)TWO_B * D_DIM * 4);
    float* sum_es = sum_e + TWO_B;
    float* pos    = sum_es + TWO_B;

    norm_split_kernel<<<TWO_B, 256, 0, stream>>>(features, fh, fl);
    zero_kernel<<<(3 * TWO_B + 255) / 256, 256, 0, stream>>>(sum_e, 3 * TWO_B);
    fused_sim_kernel<<<dim3(TWO_B / MB, 8, 1), 256, 0, stream>>>(
        fh, fl, tau_buf, index, sum_e, sum_es, pos);
    copy_out_kernel<<<4096, 256, 0, stream>>>(s_buf, u_buf, tau_buf, out, N);
    finalize_kernel<<<1, 1024, 0, stream>>>(sum_e, sum_es, pos,
                                            s_buf, tau_buf, u_buf,
                                            index, epoch, out, N);
}
